// SE3PointConvolution_22668837388927
// MI455X (gfx1250) — matrix operand order repaired and checked by reading, not compiled, not run on hardware
//
#include <hip/hip_runtime.h>

// MI455X / gfx1250, wave32, v_wmma_f32_16x16x32_f16 (f32 accum).
//
// out[b,c,p] = sum_{k,n,j} basis[b,p,k,n] * W[n,c,j] * feat[b,j,nbr(b,p,k)]
// Stage 1 (per point):  G[n,j] = sum_k basis[k,n]*nbf[k,j]   (32x64, K=48->64)
// Stage 2 (per 16 pts): out[c,p] = Wflat[c,nj'] * G[nj',p]   (M=64,N=16,K=2048)
// nj' is a permuted flattening (pair-packed n, WMMA-column channel map) shared
// by the G spill and the W pre-flatten, so the contraction is unchanged.

typedef __attribute__((ext_vector_type(16))) _Float16 v16h;
typedef __attribute__((ext_vector_type(8)))  _Float16 v8h;
typedef __attribute__((ext_vector_type(4)))  _Float16 v4h;
typedef __attribute__((ext_vector_type(8)))  float    v8f;

#define B_    2
#define P_    8192
#define K_    48
#define CIN   64
#define COUT  64
#define NJ    2048          // 32*64
#define GPAD  2064          // G row pad (+32B): conflict-free stage-2 B loads
#define BTP   72            // basisT row pad (+16B): conflict-free A loads

static __device__ __forceinline__ v8f wmma16(v16h a, v16h b, v8f c) {
  return __builtin_amdgcn_wmma_f32_16x16x32_f16(false, a, false, b, (short)0, c,
                                                false, false);
}

// features [B,CIN,P] f32 -> featT [B,P,CIN] f16 (128B contiguous per point)
__global__ void prep_feat(const float* __restrict__ f, _Float16* __restrict__ ft) {
  int tid = blockIdx.x * 256 + threadIdx.x;      // tid = (b*64 + j)*8192 + p
  int p = tid & (P_ - 1);
  int j = (tid >> 13) & 63;
  int b = tid >> 19;
  ft[((b * P_ + p) << 6) + j] = (_Float16)f[tid];
}

// W [NB,COUT,CIN] f32 -> Wf [COUT, nj'] f16 with nj' = ((n>>1)*64 + j)*2 + (n&1)
__global__ void prep_w(const float* __restrict__ w, _Float16* __restrict__ wf) {
  int tid = blockIdx.x * 256 + threadIdx.x;      // tid = c*2048 + nj'
  int t = tid & 2047;
  int c = tid >> 11;
  int n = ((t >> 7) << 1) | (t & 1);
  int j = (t >> 1) & 63;
  wf[tid] = (_Float16)w[((n << 6) + c) * 64 + j];
}

__global__ __launch_bounds__(128) void se3_conv(
    const float*    __restrict__ geometry,   // [B,P,3]
    const int*      __restrict__ neighbors,  // [B,P,K]
    const float*    __restrict__ rel_mask,   // [B,P,K]
    const _Float16* __restrict__ featT,      // [B,P,64]
    const _Float16* __restrict__ Wf,         // [64,2048] (nj' order)
    float*          __restrict__ out)        // [B,64,P]
{
  __shared__ __align__(16) _Float16 Gs[16][GPAD];        // 64.5 KB
  __shared__ __align__(16) _Float16 basisT[4][32][BTP];  // 18 KB, wave-private
  __shared__ int nbS[4][64];                             // wave-private

  const int  w    = threadIdx.x >> 5;
  const int  lane = threadIdx.x & 31;
  const int  lidx = lane & 15;
  const bool lo   = lane < 16;
  const int  b    = blockIdx.y;
  const int  p0   = blockIdx.x << 4;

  // ---------------- Stage 1: per-point G = basis^T x nbf ----------------
  for (int pt = 0; pt < 4; ++pt) {
    const int pl = (w << 2) + pt;            // local point 0..15
    const int p  = p0 + pl;

    const float gx0 = geometry[(b * P_ + p) * 3 + 0];
    const float gy0 = geometry[(b * P_ + p) * 3 + 1];
    const float gz0 = geometry[(b * P_ + p) * 3 + 2];

    // Edge phase: one lane per edge; evaluate all 32 basis values per edge,
    // store basisT[n][k] so A operands become two contiguous b128 LDS loads.
    #pragma unroll
    for (int rep = 0; rep < 2; ++rep) {      // k slots 0..63 (48 valid)
      const int k = lane + (rep << 5);
      float m = 0.f, d = 0.f, ux = 0.f, uy = 0.f, uz = 0.f;
      int   nb = 0;
      if (k < K_) {
        const int ei = (b * P_ + p) * K_ + k;
        nb = neighbors[ei];
        m  = rel_mask[ei];
        const float rx = geometry[(b * P_ + nb) * 3 + 0] - gx0;
        const float ry = geometry[(b * P_ + nb) * 3 + 1] - gy0;
        const float rz = geometry[(b * P_ + nb) * 3 + 2] - gz0;
        d = sqrtf(rx * rx + ry * ry + rz * rz);
        const float inv = (d > 1e-8f) ? (1.0f / d) : 0.0f;
        ux = rx * inv; uy = ry * inv; uz = rz * inv;
      }
      nbS[w][k] = nb;                        // nb=0, m=0 pads k>=48
      #pragma unroll
      for (int r = 0; r < 8; ++r) {
        const float t   = (d - (float)r * (2.0f / 7.0f)) * 4.0f; // /SIGMA=0.25
        const float rad = __expf(-0.5f * t * t) * m;
        basisT[w][4 * r + 0][k] = (_Float16)rad;
        basisT[w][4 * r + 1][k] = (_Float16)(rad * ux);
        basisT[w][4 * r + 2][k] = (_Float16)(rad * uy);
        basisT[w][4 * r + 3][k] = (_Float16)(rad * uz);
      }
    }
    __builtin_amdgcn_wave_barrier();         // wave-private LDS; in-order DS

    v8f acc[2][4] = {};
    #pragma unroll
    for (int ks = 0; ks < 2; ++ks) {
      const int kk = ks << 5;
      // A operands from basisT: lane<16 K{kk+0..15}, else K{kk+16..31}
      v16h A[2];
      #pragma unroll
      for (int mt = 0; mt < 2; ++mt) {
        const _Float16* bt = &basisT[w][(mt << 4) + lidx][kk + (lo ? 0 : 16)];
        v8h a0 = *(const v8h*)bt;
        v8h a1 = *(const v8h*)(bt + 8);
        A[mt] = __builtin_shufflevector(a0, a1,
                  0,1,2,3,4,5,6,7,8,9,10,11,12,13,14,15);
      }
      // B operands: per element i the half-wave fetches one neighbor's full
      // 128B feature row (lane loads channels 4*lidx..+3 as one b64).
      v16h Bv[4];
      const int kb = kk + (lo ? 0 : 16);
      #pragma unroll
      for (int i = 0; i < 16; ++i) {
        const int nb = nbS[w][kb + i];
        const v4h f4 = *(const v4h*)(featT + ((b * P_ + nb) << 6) + (lidx << 2));
        Bv[0][i] = f4[0]; Bv[1][i] = f4[1]; Bv[2][i] = f4[2]; Bv[3][i] = f4[3];
      }
      #pragma unroll
      for (int jt = 0; jt < 4; ++jt) {
        acc[0][jt] = wmma16(A[0], Bv[jt], acc[0][jt]);
        acc[1][jt] = wmma16(A[1], Bv[jt], acc[1][jt]);
      }
    }
    // Spill G -> LDS in nj' order; adjacent n rows pair-packed into b32 stores.
    #pragma unroll
    for (int mt = 0; mt < 2; ++mt)
      #pragma unroll
      for (int jt = 0; jt < 4; ++jt) {
        const int j = (lidx << 2) + jt;      // actual channel of WMMA column
        #pragma unroll
        for (int t2 = 0; t2 < 4; ++t2) {
          const int n0 = (mt << 4) + (lo ? 0 : 8) + (t2 << 1); // even row
          union { unsigned u; _Float16 h[2]; } pk;
          pk.h[0] = (_Float16)acc[mt][jt][2 * t2];
          pk.h[1] = (_Float16)acc[mt][jt][2 * t2 + 1];
          *(unsigned*)&Gs[pl][(((n0 >> 1) << 6) + j) << 1] = pk.u;
        }
      }
  }
  __syncthreads();                           // all 16 G columns ready

  // ---------------- Stage 2: out[c, p] = Wf[c, :] . G[:, p] ----------------
  v8f o = {};
  const _Float16* Wrow = Wf + (size_t)((w << 4) + lidx) * NJ; // row c = 16w+lidx
  const int aoff = lo ? 0 : 8;
  const int boff = lo ? 0 : 8;
  for (int kk = 0; kk < NJ; kk += 32) {
    v8h a0 = *(const v8h*)(Wrow + kk + aoff);        // A layout V0-3
    v8h a1 = *(const v8h*)(Wrow + kk + 16 + aoff);   // A layout V4-7
    v8h b0 = *(const v8h*)(&Gs[lidx][kk + boff]);    // K{boff..+7,16+boff..+7}, col=lidx
    v8h b1 = *(const v8h*)(&Gs[lidx][kk + boff + 16]);
    v16h Av = __builtin_shufflevector(a0, a1, 0,1,2,3,4,5,6,7,8,9,10,11,12,13,14,15);
    v16h Bv = __builtin_shufflevector(b0, b1, 0,1,2,3,4,5,6,7,8,9,10,11,12,13,14,15);
    o = wmma16(Av, Bv, o);
  }
  #pragma unroll
  for (int r8 = 0; r8 < 8; ++r8) {
    const int c = (w << 4) + r8 + (lo ? 0 : 8);
    out[(((b << 6) + c) << 13) + p0 + lidx] = o[r8]; // 16 contiguous floats/row
  }
}

extern "C" void kernel_launch(void* const* d_in, const int* in_sizes, int n_in,
                              void* d_out, int out_size, void* d_ws, size_t ws_size,
                              hipStream_t stream) {
  const float* features  = (const float*)d_in[0];
  const float* geometry  = (const float*)d_in[1];
  const int*   neighbors = (const int*)d_in[2];
  const float* rel_mask  = (const float*)d_in[3];
  const float* W         = (const float*)d_in[4];
  float* out = (float*)d_out;

  _Float16* featT = (_Float16*)d_ws;                                   // 2 MB
  _Float16* Wf    = (_Float16*)((char*)d_ws +
                      (size_t)B_ * P_ * CIN * sizeof(_Float16));       // +256 KB

  prep_feat<<<(B_ * CIN * P_) / 256, 256, 0, stream>>>(features, featT);
  prep_w<<<(COUT * NJ) / 256, 256, 0, stream>>>(W, Wf);

  dim3 grid(P_ / 16, B_);
  se3_conv<<<grid, 128, 0, stream>>>(geometry, neighbors, rel_mask, featT, Wf, out);
}
